// PointnetSAModule_73065983639955
// MI455X (gfx1250) — matrix operand order repaired and checked by reading, not compiled, not run on hardware
//
#include <hip/hip_runtime.h>
#include <hip/hip_bf16.h>

#define B_ 8
#define N_ 8192
#define C_ 64
#define NPOINT_ 2048
#define NSAMPLE_ 32
#define RADIUS2_ 0.04f
#define EPS_ 1e-5f

typedef __attribute__((ext_vector_type(16))) _Float16 v16h;
typedef __attribute__((ext_vector_type(8)))  _Float16 v8h;
typedef __attribute__((ext_vector_type(8)))  float    v8f;

// ---------------------------------------------------------------------------
// Kernel 1: farthest point sampling. One workgroup per batch, 1024 threads,
// each thread owns 8 points in registers. 2048 sequential iterations of
// (dist update -> wave32 shuffle argmax -> LDS argmax across 32 waves).
// ---------------------------------------------------------------------------
__global__ __launch_bounds__(1024) void fps_kernel(const float* __restrict__ xyz,
                                                   int* __restrict__ fpsIdx) {
  const int b = blockIdx.x;
  const int t = threadIdx.x;
  const int lane = t & 31, wid = t >> 5;
  __shared__ float sC[3];
  __shared__ int sFar;
  __shared__ float sV[32];
  __shared__ int sI[32];

  float px[8], py[8], pz[8], d[8];
  const float* base = xyz + (b * N_ + t * 8) * 3;
#pragma unroll
  for (int j = 0; j < 8; ++j) {
    px[j] = base[j * 3 + 0];
    py[j] = base[j * 3 + 1];
    pz[j] = base[j * 3 + 2];
    d[j] = 1e10f;
  }
  if (t == 0) { sFar = 0; sC[0] = px[0]; sC[1] = py[0]; sC[2] = pz[0]; }
  __syncthreads();

  int far = 0;
  for (int it = 0; it < NPOINT_; ++it) {
    if (t == 0) fpsIdx[b * NPOINT_ + it] = far;
    const float cx = sC[0], cy = sC[1], cz = sC[2];
    float bv = -1.f;
    int bi = t * 8;
#pragma unroll
    for (int j = 0; j < 8; ++j) {
      float dx = px[j] - cx, dy = py[j] - cy, dz = pz[j] - cz;
      float dd = dx * dx + dy * dy + dz * dz;
      d[j] = fminf(d[j], dd);
      if (d[j] > bv) { bv = d[j]; bi = t * 8 + j; }
    }
#pragma unroll
    for (int off = 16; off > 0; off >>= 1) {  // wave32 argmax (lowest idx wins ties)
      float ov = __shfl_xor(bv, off, 32);
      int   oi = __shfl_xor(bi, off, 32);
      if (ov > bv || (ov == bv && oi < bi)) { bv = ov; bi = oi; }
    }
    if (lane == 0) { sV[wid] = bv; sI[wid] = bi; }
    __syncthreads();
    if (wid == 0) {
      float v2 = sV[lane];
      int i2 = sI[lane];
#pragma unroll
      for (int off = 16; off > 0; off >>= 1) {
        float ov = __shfl_xor(v2, off, 32);
        int   oi = __shfl_xor(i2, off, 32);
        if (ov > v2 || (ov == v2 && oi < i2)) { v2 = ov; i2 = oi; }
      }
      if (lane == 0) sFar = i2;
    }
    __syncthreads();
    far = sFar;
    if ((far >> 3) == t) {  // owner of winning point publishes the new centroid
      int j = far & 7;
      sC[0] = px[j]; sC[1] = py[j]; sC[2] = pz[j];
    }
    __syncthreads();
  }
}

// ---------------------------------------------------------------------------
// Kernel 2: gather new_xyz = xyz[fps_idx] straight into d_out[0 .. 49151].
// ---------------------------------------------------------------------------
__global__ __launch_bounds__(256) void gather_newxyz_kernel(const float* __restrict__ xyz,
                                                            const int* __restrict__ fpsIdx,
                                                            float* __restrict__ out) {
  int i = blockIdx.x * 256 + threadIdx.x;
  if (i >= B_ * NPOINT_ * 3) return;
  int b = i / (NPOINT_ * 3);
  int r = i - b * (NPOINT_ * 3);
  int s = r / 3, k = r - s * 3;
  int idx = fpsIdx[b * NPOINT_ + s];
  out[i] = xyz[(b * N_ + idx) * 3 + k];
}

// ---------------------------------------------------------------------------
// Kernel 3: ball query. One wave32 per query; ballot+popc stream-compaction
// picks the first 32 in-range indices in ascending order (== reference's
// sort-of-masked-arange), padding with the first in-range index.
// ---------------------------------------------------------------------------
__global__ __launch_bounds__(256) void ballquery_kernel(const float* __restrict__ xyz,
                                                        const float* __restrict__ newxyz,
                                                        int* __restrict__ gidx) {
  const int lane = threadIdx.x & 31;
  const int wid = threadIdx.x >> 5;
  const int q = blockIdx.x * 8 + wid;          // 16384 queries total
  const int b = q >> 11;                       // /2048
  const float qx = newxyz[q * 3 + 0];
  const float qy = newxyz[q * 3 + 1];
  const float qz = newxyz[q * 3 + 2];
  int* outq = gidx + q * NSAMPLE_;

  int cnt = 0;
  int first = -1;
  for (int basei = 0; basei < N_; basei += 32) {
    int i = basei + lane;
    const float* p = xyz + (b * N_ + i) * 3;
    float dx = p[0] - qx, dy = p[1] - qy, dz = p[2] - qz;
    bool in = (dx * dx + dy * dy + dz * dz) <= RADIUS2_;
    unsigned m = (unsigned)__ballot(in);
    if (first < 0 && m) first = basei + (__ffs(m) - 1);
    int pos = cnt + __popc(m & ((1u << lane) - 1u));
    if (in && pos < NSAMPLE_) outq[pos] = i;
    cnt += __popc(m);
    if (cnt >= NSAMPLE_) break;
  }
  if (first < 0) first = 0;
  for (int p = cnt + lane; p < NSAMPLE_; p += 32) outq[p] = first;
}

// ---------------------------------------------------------------------------
// Kernel 4: fused gather + 3-layer MLP (WMMA f16->f32) + BN + ReLU + max-pool.
// 256 threads = 8 waves per block; each wave owns one query group (32x67).
// Weights staged once per block into LDS in B-fragment-friendly [N][K] rows.
// ---------------------------------------------------------------------------

// strides in halfs; chosen so per-lane ds_load_b128 are bank-conflict free
#define XSTR 104   // 52 dwords, gcd(52,64)=4 -> perfect 64-bank spread for b128
#define YSTR 72    // 36 dwords, same property

__device__ __forceinline__ v16h cat8(v8h lo, v8h hi) {
  v16h r;
#pragma unroll
  for (int i = 0; i < 8; ++i) { r[i] = lo[i]; r[i + 8] = hi[i]; }
  return r;
}

// A-matrix 16x32 f16 fragment (ISA layout): lane<16 holds row M=mbase+lane,
// K = {k0..k0+7, k0+16..k0+23}; lane>=16 same row group offset by +8 in K.
__device__ __forceinline__ v16h load_a_frag(const _Float16* __restrict__ buf, int stride,
                                            int mbase, int k0, int lane) {
  int row = mbase + (lane & 15);
  int koff = k0 + ((lane & 16) ? 8 : 0);
  const _Float16* p = buf + row * stride + koff;
  v8h lo = *(const v8h*)(p);
  v8h hi = *(const v8h*)(p + 16);
  return cat8(lo, hi);
}

// B-matrix 32x16 f16 fragment: lane holds column N=n0+(lane&15),
// K = {k0..k0+7, k0+16..k0+23}; lane>=16 offset by +8 in K (matches A).
// Weights are stored [N][K] row-major, so this is two 16-byte loads.
__device__ __forceinline__ v16h load_b_frag(const _Float16* __restrict__ wt, int stride,
                                            int n0, int k0, int lane) {
  int row = n0 + (lane & 15);
  int koff = k0 + ((lane & 16) ? 8 : 0);
  const _Float16* p = wt + row * stride + koff;
  v8h lo = *(const v8h*)(p);
  v8h hi = *(const v8h*)(p + 16);
  return cat8(lo, hi);
}

// D layout: acc[v] = element (M = mbase + v + (lane>=16 ? 8 : 0), N = n0+(lane&15)).
__device__ __forceinline__ void store_d_bnrelu(_Float16* __restrict__ buf, int stride,
                                               int mbase, int n0, int lane, v8f acc,
                                               float scale, float shift) {
  int col = n0 + (lane & 15);
  int rbase = mbase + ((lane & 16) ? 8 : 0);
#pragma unroll
  for (int v = 0; v < 8; ++v) {
    float y = scale * acc[v] + shift;
    y = y > 0.f ? y : 0.f;
    buf[(rbase + v) * stride + col] = (_Float16)y;
  }
}

// Fold bias + batchnorm into y = scale*x + shift (per output channel).
__device__ __forceinline__ void bn_coef(const float* __restrict__ bias,
                                        const float* __restrict__ g,
                                        const float* __restrict__ be,
                                        const float* __restrict__ m,
                                        const float* __restrict__ v,
                                        int ch, float& scale, float& shift) {
  float rs = rsqrtf(v[ch] + EPS_);
  float sc = g[ch] * rs;
  scale = sc;
  shift = sc * (bias[ch] - m[ch]) + be[ch];
}

#define W0_ELEMS (64 * XSTR)
#define W1_ELEMS (64 * YSTR)
#define W2_ELEMS (128 * YSTR)
#define XB_ELEMS (32 * XSTR)
#define YB_ELEMS (32 * YSTR)
#define MLP_LDS_HALFS (W0_ELEMS + W1_ELEMS + W2_ELEMS + 8 * XB_ELEMS + 8 * YB_ELEMS)

__global__ __launch_bounds__(256) void mlp_kernel(
    const float* __restrict__ xyz, const float* __restrict__ features,
    const float* __restrict__ newxyz, const int* __restrict__ gidx,
    const float* __restrict__ w0, const float* __restrict__ b0,
    const float* __restrict__ g0, const float* __restrict__ be0,
    const float* __restrict__ m0, const float* __restrict__ v0,
    const float* __restrict__ w1, const float* __restrict__ b1,
    const float* __restrict__ g1, const float* __restrict__ be1,
    const float* __restrict__ m1, const float* __restrict__ v1,
    const float* __restrict__ w2, const float* __restrict__ b2,
    const float* __restrict__ g2, const float* __restrict__ be2,
    const float* __restrict__ m2, const float* __restrict__ v2,
    float* __restrict__ outF) {
  extern __shared__ _Float16 smem[];
  _Float16* W0s = smem;                       // [64][XSTR], cols 67..95 zero
  _Float16* W1s = W0s + W0_ELEMS;             // [64][YSTR]
  _Float16* W2s = W1s + W1_ELEMS;             // [128][YSTR]
  _Float16* Xall = W2s + W2_ELEMS;
  _Float16* Yall = Xall + 8 * XB_ELEMS;

  const int tid = threadIdx.x;
  const int lane = tid & 31;
  const int wid = tid >> 5;

  // --- stage weights into LDS as f16 [N][K] rows (cooperative) ---
  for (int i = tid; i < 64 * 96; i += 256) {        // W0: 64x67, K padded to 96
    int r = i / 96, c = i - r * 96;
    float val = (c < 67) ? w0[r * 67 + c] : 0.f;
    W0s[r * XSTR + c] = (_Float16)val;
  }
  for (int i = tid; i < 64 * 64; i += 256) {        // W1: 64x64
    int r = i >> 6, c = i & 63;
    W1s[r * YSTR + c] = (_Float16)w1[r * 64 + c];
  }
  for (int i = tid; i < 128 * 64; i += 256) {       // W2: 128x64
    int r = i >> 6, c = i & 63;
    W2s[r * YSTR + c] = (_Float16)w2[r * 64 + c];
  }
  __syncthreads();

  // --- each wave handles one query group ---
  const int q = blockIdx.x * 8 + wid;          // 0 .. 16383
  const int b = q >> 11;
  const int s = q & 2047;
  _Float16* Xb = Xall + wid * XB_ELEMS;        // layer0 input [32][XSTR]
  _Float16* Yb = Yall + wid * YB_ELEMS;        // layer1 input [32][YSTR]

  // zero the K-pad region read by the k0=64 tile (cols 67..95)
  {
    _Float16* row = Xb + lane * XSTR;
#pragma unroll
    for (int c = 67; c < 96; ++c) row[c] = (_Float16)0.f;
  }

  // gather: lane = sample j
  {
    int nidx = gidx[q * NSAMPLE_ + lane];
    const float* p = xyz + (b * N_ + nidx) * 3;
    float cx = newxyz[q * 3 + 0], cy = newxyz[q * 3 + 1], cz = newxyz[q * 3 + 2];
    _Float16* row = Xb + lane * XSTR;
    row[0] = (_Float16)(p[0] - cx);
    row[1] = (_Float16)(p[1] - cy);
    row[2] = (_Float16)(p[2] - cz);
    const float* fb = features + b * (C_ * N_) + nidx;
#pragma unroll 8
    for (int c = 0; c < C_; ++c) row[3 + c] = (_Float16)fb[c * N_];
  }

  // ---------------- layer 0: [32x96] x [96x64] ----------------
  {
    v16h a[2][3];
#pragma unroll
    for (int mt = 0; mt < 2; ++mt)
#pragma unroll
      for (int k = 0; k < 3; ++k) a[mt][k] = load_a_frag(Xb, XSTR, mt * 16, k * 32, lane);
#pragma unroll
    for (int n = 0; n < 4; ++n) {
      float scale, shift;
      bn_coef(b0, g0, be0, m0, v0, n * 16 + (lane & 15), scale, shift);
      v8f acc0 = {};
      v8f acc1 = {};
#pragma unroll
      for (int k = 0; k < 3; ++k) {
        v16h bf = load_b_frag(W0s, XSTR, n * 16, k * 32, lane);
        acc0 = __builtin_amdgcn_wmma_f32_16x16x32_f16(false, a[0][k], false, bf, (short)0, acc0, false, false);
        acc1 = __builtin_amdgcn_wmma_f32_16x16x32_f16(false, a[1][k], false, bf, (short)0, acc1, false, false);
      }
      store_d_bnrelu(Yb, YSTR, 0, n * 16, lane, acc0, scale, shift);
      store_d_bnrelu(Yb, YSTR, 16, n * 16, lane, acc1, scale, shift);
    }
  }

  // ---------------- layer 1: [32x64] x [64x64] -> Xb (stride YSTR) ---------
  {
    v16h a[2][2];
#pragma unroll
    for (int mt = 0; mt < 2; ++mt)
#pragma unroll
      for (int k = 0; k < 2; ++k) a[mt][k] = load_a_frag(Yb, YSTR, mt * 16, k * 32, lane);
#pragma unroll
    for (int n = 0; n < 4; ++n) {
      float scale, shift;
      bn_coef(b1, g1, be1, m1, v1, n * 16 + (lane & 15), scale, shift);
      v8f acc0 = {};
      v8f acc1 = {};
#pragma unroll
      for (int k = 0; k < 2; ++k) {
        v16h bf = load_b_frag(W1s, YSTR, n * 16, k * 32, lane);
        acc0 = __builtin_amdgcn_wmma_f32_16x16x32_f16(false, a[0][k], false, bf, (short)0, acc0, false, false);
        acc1 = __builtin_amdgcn_wmma_f32_16x16x32_f16(false, a[1][k], false, bf, (short)0, acc1, false, false);
      }
      store_d_bnrelu(Xb, YSTR, 0, n * 16, lane, acc0, scale, shift);
      store_d_bnrelu(Xb, YSTR, 16, n * 16, lane, acc1, scale, shift);
    }
  }

  // ---------------- layer 2: [32x64] x [64x128] + max over 32 samples ------
  {
    v16h a[2][2];
#pragma unroll
    for (int mt = 0; mt < 2; ++mt)
#pragma unroll
      for (int k = 0; k < 2; ++k) a[mt][k] = load_a_frag(Xb, YSTR, mt * 16, k * 32, lane);
#pragma unroll
    for (int n = 0; n < 8; ++n) {
      float scale, shift;
      bn_coef(b2, g2, be2, m2, v2, n * 16 + (lane & 15), scale, shift);
      v8f acc0 = {};
      v8f acc1 = {};
#pragma unroll
      for (int k = 0; k < 2; ++k) {
        v16h bf = load_b_frag(W2s, YSTR, n * 16, k * 32, lane);
        acc0 = __builtin_amdgcn_wmma_f32_16x16x32_f16(false, a[0][k], false, bf, (short)0, acc0, false, false);
        acc1 = __builtin_amdgcn_wmma_f32_16x16x32_f16(false, a[1][k], false, bf, (short)0, acc1, false, false);
      }
      float mx = -1e30f;
#pragma unroll
      for (int v = 0; v < 8; ++v) {
        float y0 = scale * acc0[v] + shift; y0 = y0 > 0.f ? y0 : 0.f;
        float y1 = scale * acc1[v] + shift; y1 = y1 > 0.f ? y1 : 0.f;
        mx = fmaxf(mx, fmaxf(y0, y1));
      }
      mx = fmaxf(mx, __shfl_xor(mx, 16, 32));  // merge the two lane halves (M rows)
      if (lane < 16) {
        int ch = n * 16 + lane;
        outF[(b * 128 + ch) * NPOINT_ + s] = mx;  // (B,128,2048) layout
      }
    }
  }
}

// ---------------------------------------------------------------------------
extern "C" void kernel_launch(void* const* d_in, const int* in_sizes, int n_in,
                              void* d_out, int out_size, void* d_ws, size_t ws_size,
                              hipStream_t stream) {
  const float* xyz = (const float*)d_in[0];
  const float* features = (const float*)d_in[1];
  const float* w0 = (const float*)d_in[2];
  const float* b0 = (const float*)d_in[3];
  const float* g0 = (const float*)d_in[4];
  const float* be0 = (const float*)d_in[5];
  const float* m0 = (const float*)d_in[6];
  const float* v0 = (const float*)d_in[7];
  const float* w1 = (const float*)d_in[8];
  const float* b1 = (const float*)d_in[9];
  const float* g1 = (const float*)d_in[10];
  const float* be1 = (const float*)d_in[11];
  const float* m1 = (const float*)d_in[12];
  const float* v1 = (const float*)d_in[13];
  const float* w2 = (const float*)d_in[14];
  const float* b2 = (const float*)d_in[15];
  const float* g2 = (const float*)d_in[16];
  const float* be2 = (const float*)d_in[17];
  const float* m2 = (const float*)d_in[18];
  const float* v2 = (const float*)d_in[19];

  float* out = (float*)d_out;                 // [0..49151]=new_xyz, rest=(B,128,2048)
  int* fpsIdx = (int*)d_ws;                   // B*NPOINT ints
  int* gidx = fpsIdx + B_ * NPOINT_;          // B*NPOINT*NSAMPLE ints

  fps_kernel<<<dim3(B_), dim3(1024), 0, stream>>>(xyz, fpsIdx);

  int nNew = B_ * NPOINT_ * 3;
  gather_newxyz_kernel<<<dim3((nNew + 255) / 256), dim3(256), 0, stream>>>(xyz, fpsIdx, out);

  ballquery_kernel<<<dim3(B_ * NPOINT_ / 8), dim3(256), 0, stream>>>(xyz, out, gidx);

  size_t ldsBytes = (size_t)MLP_LDS_HALFS * sizeof(_Float16);  // 131072 B
  mlp_kernel<<<dim3(B_ * NPOINT_ / 8), dim3(256), ldsBytes, stream>>>(
      xyz, features, out, gidx,
      w0, b0, g0, be0, m0, v0,
      w1, b1, g1, be1, m1, v1,
      w2, b2, g2, be2, m2, v2,
      out + B_ * NPOINT_ * 3);
}
